// Irregular_decoder_82025285419108
// MI455X (gfx1250) — compile-verified, matrix operand order checked by reading, not run on hardware
//
#include <hip/hip_runtime.h>
#include <cstdint>
#include <cstddef>

// ---------------------------------------------------------------------------
// TLSTM (2-layer, time-aware) for MI455X / gfx1250.
// bf16 WMMA (16x16x32) GEMMs, fp32 accumulation; weights pre-packed once into
// fragment-native layout; per-step fused cell kernel. W_comb = hl_W@Wx0 + Wh0
// folds the input projection off the 256-step critical path.
// Grid per step: (64 n-tiles, 2 m-halves) x 160 threads (4 gate waves + 1
// decay-GEMM wave). Low register pressure keeps accumulators pinned (no
// v_mov/v_nop hazard chains).
// ---------------------------------------------------------------------------

typedef __attribute__((ext_vector_type(16))) __bf16 v16bf;
typedef __attribute__((ext_vector_type(8)))  __bf16 v8bf;
typedef __attribute__((ext_vector_type(8)))  float  v8f;

union BF16Frag { v16bf v; struct { v8bf lo, hi; } s; };

#define BSZ  64
#define SEQL 256
#define HID  1024

// ---------------------------------------------------------------------------
// One GEMM term for 2 M-tiles of 16: acc[mt] += A[32,K] @ Wp(:, nt-tile).
// A: row-major bf16 (already offset to this block's 32-row half).
// Wp packed: tile (kt, nt) at (kt*ntiles + nt)*512 elems; chunk0 = lane-major
// first 8 values (16B/lane), chunk1 at +256 elems.
// ---------------------------------------------------------------------------
__device__ __forceinline__ void gemm_term(const __bf16* __restrict__ A, int lda,
                                          const __bf16* __restrict__ Wp,
                                          int nkt, int ntiles, int nt,
                                          int lane, v8f acc[2])
{
    const int r  = lane & 15;
    const int hb = (lane >> 4) & 1;          // K-half select per ISA layout
    const __bf16* a0 = A + (size_t)r * lda + hb * 8;   // M-tile 0 row
    const __bf16* a1 = a0 + (size_t)16 * lda;          // M-tile 1 row
    const __bf16* bt = Wp + (size_t)nt * 512 + lane * 8;
    const size_t  bstep = (size_t)ntiles * 512;
#pragma unroll 1
    for (int kt = 0; kt < nkt; ++kt) {
        BF16Frag bf, af0, af1;
        bf.s.lo  = *(const v8bf*)(bt);
        bf.s.hi  = *(const v8bf*)(bt + 256);
        af0.s.lo = *(const v8bf*)(a0);        // K = kb + hb*8 .. +7
        af0.s.hi = *(const v8bf*)(a0 + 16);   // K = kb + 16 + hb*8 .. +7
        af1.s.lo = *(const v8bf*)(a1);
        af1.s.hi = *(const v8bf*)(a1 + 16);
        __builtin_prefetch(bt + bstep, 0, 3);  // next K-tile of B, near scope
        acc[0] = __builtin_amdgcn_wmma_f32_16x16x32_bf16(
            false, af0.v, false, bf.v, (short)0, acc[0], false, false);
        acc[1] = __builtin_amdgcn_wmma_f32_16x16x32_bf16(
            false, af1.v, false, bf.v, (short)0, acc[1], false, false);
        bt += bstep;
        a0 += 32;
        a1 += 32;
    }
}

// ---------------------------------------------------------------------------
// Fused TLSTM cell step.
//   z    = A1@W1 (+ A2@W2) + bias            [64, 4096]
//   cs   = tanh(c_bf16@Wd + bd)              [64, 1024]
//   cadj = c - cs + cs / log(e + dt)
//   c'   = sig(f)*cadj + sig(i)*tanh(g);  h' = sig(o)*tanh(c')
// ---------------------------------------------------------------------------
__global__ __launch_bounds__(160)
void tlstm_step(const __bf16* __restrict__ A1, const __bf16* __restrict__ W1p,
                int nkt1, int lda1,
                const __bf16* __restrict__ A2, const __bf16* __restrict__ W2p,
                int nkt2,
                const __bf16* __restrict__ Wdp, const __bf16* __restrict__ Cb_in,
                const float* __restrict__ bias, const float* __restrict__ bd,
                float* __restrict__ Cf,          // [64,1024] fp32, in-place
                __bf16* __restrict__ Cb_out,     // [64,1024] bf16 ping-pong
                const float* __restrict__ dt, int dt_stride,
                __bf16* __restrict__ Hb_out,     // [64,1024] bf16
                float* __restrict__ Hf_out, long hf_row_stride)
{
    __shared__ float sz[5][32][16];
    const int wave = threadIdx.x >> 5;
    const int lane = threadIdx.x & 31;
    const int mb   = blockIdx.y;               // 0/1: rows 0-31 or 32-63

    v8f acc[2] = {};
    if (wave < 4) {
        const int nt = wave * 64 + blockIdx.x;  // gate tile within N=4096
        gemm_term(A1 + (size_t)mb * 32 * lda1, lda1, W1p, nkt1, 256, nt, lane, acc);
        if (A2)
            gemm_term(A2 + (size_t)mb * 32 * HID, HID, W2p, nkt2, 256, nt, lane, acc);
    } else {
        gemm_term(Cb_in + (size_t)mb * 32 * HID, HID, Wdp, 32, 64, blockIdx.x,
                  lane, acc);
    }

    // D layout: lane holds col = lane&15, rows v + 8*(lane>>4) per VGPR v.
    const int col = lane & 15;
    const int hi8 = ((lane >> 4) & 1) * 8;
#pragma unroll
    for (int mt = 0; mt < 2; ++mt)
#pragma unroll
        for (int v = 0; v < 8; ++v)
            sz[wave][mt * 16 + hi8 + v][col] = acc[mt][v];
    __syncthreads();

    const int n0 = blockIdx.x * 16;
    for (int idx = threadIdx.x; idx < 32 * 16; idx += 160) {
        const int ml = idx >> 4, jl = idx & 15;
        const int m = mb * 32 + ml, j = n0 + jl;
        const float zi = sz[0][ml][jl] + bias[j];
        const float zf = sz[1][ml][jl] + bias[1024 + j];
        const float zg = sz[2][ml][jl] + bias[2048 + j];
        const float zo = sz[3][ml][jl] + bias[3072 + j];
        const float cs = tanhf(sz[4][ml][jl] + bd[j]);
        const float dtv  = dt[(size_t)m * dt_stride];
        const float gdec = 1.0f / logf(2.718281828459045f + dtv);
        const float cold = Cf[m * HID + j];
        const float cadj = cold - cs + cs * gdec;
        const float si = 1.0f / (1.0f + expf(-zi));
        const float sf = 1.0f / (1.0f + expf(-zf));
        const float so = 1.0f / (1.0f + expf(-zo));
        const float cn = sf * cadj + si * tanhf(zg);
        const float hn = so * tanhf(cn);
        Cf[m * HID + j]     = cn;
        Cb_out[m * HID + j] = (__bf16)cn;
        Hb_out[m * HID + j] = (__bf16)hn;
        if (Hf_out) Hf_out[(size_t)m * hf_row_stride + j] = hn;
    }
}

// --------------------------- one-time prep kernels --------------------------

// W_comb[i,n] = sum_k hl_W[i,k]*Wx0[k,n] + Wh0[i,n]   (fp32)
__global__ void wcomb_kernel(const float* __restrict__ hlW,
                             const float* __restrict__ Wx0,
                             const float* __restrict__ Wh0,
                             float* __restrict__ WC)
{
    const int n = blockIdx.x * blockDim.x + threadIdx.x;  // 0..4095
    const int i = blockIdx.y;                             // 0..1023
    float s = Wh0[(size_t)i * 4096 + n];
    for (int k = 0; k < 512; ++k)
        s += hlW[(size_t)i * 512 + k] * Wx0[(size_t)k * 4096 + n];
    WC[(size_t)i * 4096 + n] = s;
}

// b_comb[n] = sum_k hl_b[k]*Wx0[k,n] + b0[n]
__global__ void bcomb_kernel(const float* __restrict__ hlb,
                             const float* __restrict__ Wx0,
                             const float* __restrict__ b0,
                             float* __restrict__ bc)
{
    const int n = blockIdx.x * blockDim.x + threadIdx.x;
    float s = b0[n];
    for (int k = 0; k < 512; ++k) s += hlb[k] * Wx0[(size_t)k * 4096 + n];
    bc[n] = s;
}

// fp32 [K,N] row-major -> fragment-native packed bf16.
__global__ void pack_kernel(const float* __restrict__ W, __bf16* __restrict__ dst,
                            int N, int ntiles, long total)
{
    const long idx = (long)blockIdx.x * 256 + threadIdx.x;
    if (idx >= total) return;
    const long tile = idx >> 9;
    const int  r = (int)(idx & 511);
    const int  p = r >> 8, rr = r & 255;
    const int  lane = rr >> 3, e = rr & 7;
    const int  nt = (int)(tile % ntiles);
    const long kt = tile / ntiles;
    const int  n = nt * 16 + (lane & 15);
    const int  klocal = p * 16 + ((lane < 16) ? 0 : 8) + e;
    const long k = kt * 32 + klocal;
    dst[idx] = (__bf16)W[k * (long)N + n];
}

// x0 = last_input @ hl_W + hl_b   -> bf16 [64,512]   (step-0 only)
__global__ void x0_kernel(const float* __restrict__ last_in,
                          const float* __restrict__ hlW,
                          const float* __restrict__ hlb,
                          __bf16* __restrict__ x0)
{
    const int n = blockIdx.x * blockDim.x + threadIdx.x;  // 0..511
    const int m = blockIdx.y;                             // 0..63
    float s = hlb[n];
    for (int k = 0; k < 1024; ++k) s += last_in[m * 1024 + k] * hlW[(size_t)k * 512 + n];
    x0[m * 512 + n] = (__bf16)s;
}

__global__ void cvt_kernel(const float* __restrict__ src, __bf16* db, float* df, int n)
{
    const int i = blockIdx.x * blockDim.x + threadIdx.x;
    if (i < n) { const float v = src[i]; if (db) db[i] = (__bf16)v; if (df) df[i] = v; }
}

__global__ void zero_bf16(__bf16* p, int n)
{ const int i = blockIdx.x * blockDim.x + threadIdx.x; if (i < n) p[i] = (__bf16)0.0f; }

__global__ void zero_f32(float* p, int n)
{ const int i = blockIdx.x * blockDim.x + threadIdx.x; if (i < n) p[i] = 0.0f; }

// hT = seq1[:, S-1, :]; cT = c1
__global__ void finalize_kernel(const float* __restrict__ seq_last, long row_stride,
                                const float* __restrict__ c1,
                                float* __restrict__ hT, float* __restrict__ cT)
{
    const int i = blockIdx.x * blockDim.x + threadIdx.x;  // 0..65535
    const int m = i >> 10, j = i & 1023;
    hT[i] = seq_last[(size_t)m * row_stride + j];
    cT[i] = c1[i];
}

// ---------------------------------------------------------------------------

extern "C" void kernel_launch(void* const* d_in, const int* in_sizes, int n_in,
                              void* d_out, int out_size, void* d_ws, size_t ws_size,
                              hipStream_t stream)
{
    const float* init_h  = (const float*)d_in[0];
    const float* init_c  = (const float*)d_in[1];
    const float* last_in = (const float*)d_in[2];
    const float* tdelta  = (const float*)d_in[3];
    const float* hl_W    = (const float*)d_in[4];
    const float* hl_b    = (const float*)d_in[5];
    const float* Wx0     = (const float*)d_in[6];
    const float* Wh0     = (const float*)d_in[7];
    const float* b0      = (const float*)d_in[8];
    const float* Wd0     = (const float*)d_in[9];
    const float* bd0     = (const float*)d_in[10];
    const float* Wx1     = (const float*)d_in[11];
    const float* Wh1     = (const float*)d_in[12];
    const float* b1      = (const float*)d_in[13];
    const float* Wd1     = (const float*)d_in[14];
    const float* bd1     = (const float*)d_in[15];

    uint8_t* p = (uint8_t*)d_ws;
    auto carve = [&](size_t bytes) -> uint8_t* {
        uint8_t* r = p; p += (bytes + 255) & ~(size_t)255; return r;
    };
    float*  WCf    = (float*)carve((size_t)1024 * 4096 * 4);
    float*  bcomb  = (float*)carve(4096 * 4);
    __bf16* WC_p   = (__bf16*)carve((size_t)1024 * 4096 * 2);
    __bf16* Wx0_p  = (__bf16*)carve((size_t)512  * 4096 * 2);
    __bf16* Wh0_p  = (__bf16*)carve((size_t)1024 * 4096 * 2);
    __bf16* Wd0_p  = (__bf16*)carve((size_t)1024 * 1024 * 2);
    __bf16* Wx1_p  = (__bf16*)carve((size_t)1024 * 4096 * 2);
    __bf16* Wh1_p  = (__bf16*)carve((size_t)1024 * 4096 * 2);
    __bf16* Wd1_p  = (__bf16*)carve((size_t)1024 * 1024 * 2);
    __bf16* x0b    = (__bf16*)carve((size_t)64 * 512 * 2);
    __bf16* h0init = (__bf16*)carve((size_t)64 * 1024 * 2);
    __bf16* seq0   = (__bf16*)carve((size_t)SEQL * 64 * 1024 * 2);
    __bf16* h1A    = (__bf16*)carve((size_t)64 * 1024 * 2);
    __bf16* h1B    = (__bf16*)carve((size_t)64 * 1024 * 2);
    __bf16* c0bA   = (__bf16*)carve((size_t)64 * 1024 * 2);
    __bf16* c0bB   = (__bf16*)carve((size_t)64 * 1024 * 2);
    __bf16* c1bA   = (__bf16*)carve((size_t)64 * 1024 * 2);
    __bf16* c1bB   = (__bf16*)carve((size_t)64 * 1024 * 2);
    float*  c0f    = (float*)carve((size_t)64 * 1024 * 4);
    float*  c1f    = (float*)carve((size_t)64 * 1024 * 4);

    // ---- one-time prep (parallel, off the recurrent critical path) ----
    wcomb_kernel<<<dim3(16, 1024), 256, 0, stream>>>(hl_W, Wx0, Wh0, WCf);
    bcomb_kernel<<<16, 256, 0, stream>>>(hl_b, Wx0, b0, bcomb);

    auto pack = [&](const float* W, __bf16* dst, int N, int K) {
        const long total = (long)K * N;
        pack_kernel<<<(int)((total + 255) / 256), 256, 0, stream>>>(W, dst, N, N / 16, total);
    };
    pack(WCf, WC_p, 4096, 1024);
    pack(Wx0, Wx0_p, 4096, 512);
    pack(Wh0, Wh0_p, 4096, 1024);
    pack(Wd0, Wd0_p, 1024, 1024);
    pack(Wx1, Wx1_p, 4096, 1024);
    pack(Wh1, Wh1_p, 4096, 1024);
    pack(Wd1, Wd1_p, 1024, 1024);

    x0_kernel<<<dim3(2, 64), 256, 0, stream>>>(last_in, hl_W, hl_b, x0b);
    cvt_kernel<<<256, 256, 0, stream>>>(init_h, h0init, nullptr, 65536);
    cvt_kernel<<<256, 256, 0, stream>>>(init_c, c0bA, c0f, 65536);
    zero_bf16<<<256, 256, 0, stream>>>(h1A, 65536);
    zero_bf16<<<256, 256, 0, stream>>>(c1bA, 65536);
    zero_f32 <<<256, 256, 0, stream>>>(c1f, 65536);

    const dim3 step_grid(64, 2);

    // ---- layer 0: 256 sequential fused-cell kernels ----
    for (int t = 0; t < SEQL; ++t) {
        const __bf16* hprev = (t == 0) ? h0init : seq0 + (size_t)(t - 1) * 65536;
        __bf16* hout = seq0 + (size_t)t * 65536;
        const __bf16* cin  = (t & 1) ? c0bB : c0bA;
        __bf16*       cout = (t & 1) ? c0bA : c0bB;
        if (t == 0) {
            // generic two-term path: z = x0@Wx0 + h@Wh0 + b0
            tlstm_step<<<step_grid, 160, 0, stream>>>(
                x0b, Wx0_p, 16, 512, h0init, Wh0_p, 32,
                Wd0_p, cin, b0, bd0, c0f, cout,
                tdelta + t, SEQL, hout, nullptr, 0);
        } else {
            // folded path: z = h@W_comb + b_comb
            tlstm_step<<<step_grid, 160, 0, stream>>>(
                hprev, WC_p, 32, 1024, nullptr, nullptr, 0,
                Wd0_p, cin, bcomb, bd0, c0f, cout,
                tdelta + t, SEQL, hout, nullptr, 0);
        }
    }

    // ---- layer 1: 256 sequential fused-cell kernels, fp32 h into d_out ----
    float* outf = (float*)d_out;
    for (int t = 0; t < SEQL; ++t) {
        const __bf16* x = seq0 + (size_t)t * 65536;
        const __bf16* hin  = (t & 1) ? h1B : h1A;
        __bf16*       hout = (t & 1) ? h1A : h1B;
        const __bf16* cin  = (t & 1) ? c1bB : c1bA;
        __bf16*       cout = (t & 1) ? c1bA : c1bB;
        tlstm_step<<<step_grid, 160, 0, stream>>>(
            x, Wx1_p, 32, 1024, hin, Wh1_p, 32,
            Wd1_p, cin, b1, bd1, c1f, cout,
            tdelta + t, SEQL, hout,
            outf + (size_t)t * 1024, (long)SEQL * 1024);
    }

    // ---- (hT, cT) tail of d_out ----
    float* hT = outf + (size_t)64 * SEQL * 1024;
    float* cT = hT + 65536;
    finalize_kernel<<<256, 256, 0, stream>>>(outf + (size_t)(SEQL - 1) * 1024,
                                             (long)SEQL * 1024, c1f, hT, cT);
}
